// GhostLinear_40518721471149
// MI455X (gfx1250) — matrix operand order repaired and checked by reading, not compiled, not run on hardware
//
#include <hip/hip_runtime.h>
#include <hip/hip_bf16.h>

// ---- problem constants (match reference) ----
#define B_   4
#define S_   2048
#define M_   (B_ * S_)   // 8192
#define K_   4096        // IN_F
#define N_   4096        // OUT_F
#define LUTN 256

// ---- GEMM tiling ----
#define BM 128
#define BN 128
#define BK 32

typedef __attribute__((ext_vector_type(16))) _Float16 v16h;
typedef __attribute__((ext_vector_type(8)))  float    v8f;
typedef __attribute__((ext_vector_type(4)))  unsigned int u32x4;
typedef __attribute__((ext_vector_type(4)))  float    f32x4;

union ABFrag { v16h h; u32x4 q[2]; };

#if __has_builtin(__builtin_amdgcn_global_load_async_to_lds_b128)
#define USE_ASYNC_LDS 1
// exact pointee type from the builtin's signature: GCC vector of 4 ints
typedef int gv4i __attribute__((vector_size(16)));
typedef __attribute__((address_space(1))) gv4i* as1_v4ip;   // global src
typedef __attribute__((address_space(3))) gv4i* as3_v4ip;   // LDS dst
#else
#define USE_ASYNC_LDS 0
#endif

__device__ __forceinline__ void wait_async0() {
#if __has_builtin(__builtin_amdgcn_s_wait_asynccnt)
    __builtin_amdgcn_s_wait_asynccnt(0);
#else
    asm volatile("s_wait_asynccnt 0x0" ::: "memory");
#endif
}

// ============================================================
// Kernel 1: dequant + transpose.
//   Wt[k][n] = (f16)( lut[idx[n][k]] * scale[n] ),  Wt is [K_][N_] f16
// ============================================================
__global__ __launch_bounds__(256) void ghost_dequant_tr(
    const int* __restrict__ idx, const float* __restrict__ scale,
    const float* __restrict__ lut, _Float16* __restrict__ Wt)
{
    __shared__ float    s_lut[LUTN];
    __shared__ _Float16 tile[32][33];   // [k_local][n_local], padded

    const int t = threadIdx.x;
    if (t < LUTN) s_lut[t] = lut[t];
    __syncthreads();

    const int n0 = blockIdx.x * 32;
    const int k0 = blockIdx.y * 32;

    const int kl = t & 31;
    const int ng = t >> 5;          // 0..7
#pragma unroll
    for (int r = 0; r < 4; ++r) {
        const int nl = r * 8 + ng;
        const int n  = n0 + nl;
        const int c  = idx[(size_t)n * K_ + (k0 + kl)] & 255;
        tile[kl][nl] = (_Float16)(s_lut[c] * scale[n]);
    }
    __syncthreads();

    const int nw = t & 31;
    const int kg = t >> 5;
#pragma unroll
    for (int r = 0; r < 4; ++r) {
        const int kw = r * 8 + kg;
        Wt[(size_t)(k0 + kw) * N_ + (n0 + nw)] = tile[kw][nw];
    }
}

// ============================================================
// Kernel 2: x f32 -> f16, 8 elements per thread, b128 stores.
// ============================================================
__global__ __launch_bounds__(256) void ghost_cvt_f16(
    const float* __restrict__ x, _Float16* __restrict__ xh)
{
    const size_t i = ((size_t)blockIdx.x * blockDim.x + threadIdx.x) * 8;
    f32x4 a = *(const f32x4*)(x + i);
    f32x4 b = *(const f32x4*)(x + i + 4);
    union { _Float16 h[8]; u32x4 q; } o;
#pragma unroll
    for (int j = 0; j < 4; ++j) o.h[j]     = (_Float16)a[j];
#pragma unroll
    for (int j = 0; j < 4; ++j) o.h[4 + j] = (_Float16)b[j];
    *(u32x4*)(xh + i) = o.q;
}

// ============================================================
// Kernel 3: GEMM  out[m][n] = sum_k Xh[m][k] * Wt[k][n]
// 256 threads (8 wave32), block tile 128x128, K-step 32.
// Wave grid 2x4; each wave 64x32 = 4x2 WMMA accumulators.
// Async path: double-buffered LDS fed by global_load_async_to_lds_b128,
// one barrier per K-step; DMA of tile k+1 overlaps WMMAs of tile k.
// ============================================================
__global__ __launch_bounds__(256) void ghost_gemm_f16(
    const _Float16* __restrict__ Xh, const _Float16* __restrict__ Wt,
    float* __restrict__ out)
{
#if USE_ASYNC_LDS
    __shared__ _Float16 lA[2][BM][BK + 8];  // 2 x 128 x 40 halfs
    __shared__ _Float16 lB[2][BK][BN + 8];  // 2 x  32 x 136 halfs
#else
    __shared__ _Float16 lA[1][BM][BK + 8];
    __shared__ _Float16 lB[1][BK][BN + 8];
#endif

    const int t     = threadIdx.x;
    const int lane  = t & 31;
    const int wave  = t >> 5;
    const int wm    = (wave >> 2) * 64;   // wave M origin (0 / 64)
    const int wn    = (wave & 3) * 32;    // wave N origin (0/32/64/96)
    const int m_blk = blockIdx.y * BM;
    const int n_blk = blockIdx.x * BN;
    const int l15   = lane & 15;
    const int h16   = lane >> 4;          // 0/1

    v8f acc[4][2];
#pragma unroll
    for (int mt = 0; mt < 4; ++mt)
#pragma unroll
        for (int nt = 0; nt < 2; ++nt)
            acc[mt][nt] = (v8f){0.f, 0.f, 0.f, 0.f, 0.f, 0.f, 0.f, 0.f};

#if USE_ASYNC_LDS
    // ---------- async DMA double-buffered pipeline ----------
    auto issue_async = [&](int buf, int k0) {
#pragma unroll
        for (int i = 0; i < 2; ++i) {
            const int c = t + i * 256;
            // A: 128 rows x 4 chunks of 8 halfs; B: 32 rows x 16 chunks
            const _Float16* gA = Xh + (size_t)(m_blk + (c >> 2)) * K_ + k0 + ((c & 3) << 3);
            const _Float16* gB = Wt + (size_t)(k0 + (c >> 4)) * N_ + n_blk + ((c & 15) << 3);
            __builtin_amdgcn_global_load_async_to_lds_b128(
                (as1_v4ip)gA, (as3_v4ip)&lA[buf][c >> 2][(c & 3) << 3], 0, 0);
            __builtin_amdgcn_global_load_async_to_lds_b128(
                (as1_v4ip)gB, (as3_v4ip)&lB[buf][c >> 4][(c & 15) << 3], 0, 0);
        }
    };

    issue_async(0, 0);
    int cur = 0;
    for (int k0 = 0; k0 < K_; k0 += BK) {
        wait_async0();        // this wave's DMA into lX[cur] done
        __syncthreads();      // all waves' DMA done; prior-buffer reads done
        if (k0 + BK < K_) issue_async(cur ^ 1, k0 + BK);   // overlap with WMMAs
#else
    // ---------- fallback: register staging, single buffer ----------
    u32x4 ra[2], rb[2];
    auto load_global = [&](int k0) {
#pragma unroll
        for (int i = 0; i < 2; ++i) {
            const int c = t + i * 256;
            ra[i] = *(const u32x4*)(Xh + (size_t)(m_blk + (c >> 2)) * K_ + k0 + ((c & 3) << 3));
            rb[i] = *(const u32x4*)(Wt + (size_t)(k0 + (c >> 4)) * N_ + n_blk + ((c & 15) << 3));
        }
    };
    load_global(0);
    const int cur = 0;
    for (int k0 = 0; k0 < K_; k0 += BK) {
        __syncthreads();
#pragma unroll
        for (int i = 0; i < 2; ++i) {
            const int c = t + i * 256;
            *(u32x4*)&lA[0][c >> 2][(c & 3) << 3]  = ra[i];
            *(u32x4*)&lB[0][c >> 4][(c & 15) << 3] = rb[i];
        }
        __syncthreads();
        if (k0 + BK < K_) load_global(k0 + BK);
#endif

        // A fragments: lane = row M (wm + mt*16 + l15);
        //   elems 0-7 = K[8*h16 .. +7], elems 8-15 = K[16+8*h16 .. +7]
        ABFrag a[4];
#pragma unroll
        for (int mt = 0; mt < 4; ++mt) {
            const int row = wm + mt * 16 + l15;
            a[mt].q[0] = *(const u32x4*)&lA[cur][row][8 * h16];
            a[mt].q[1] = *(const u32x4*)&lA[cur][row][16 + 8 * h16];
        }
        // B fragments: lane = N column (wn + nt*16 + l15);
        //   elems 0-7 = K[8*h16 .. +7], elems 8-15 = K[16+8*h16 .. +7]
        ABFrag b[2];
#pragma unroll
        for (int nt = 0; nt < 2; ++nt) {
            const int ncol = wn + nt * 16 + l15;
#pragma unroll
            for (int s = 0; s < 8; ++s) {
                b[nt].h[s]     = lB[cur][8 * h16 + s][ncol];
                b[nt].h[8 + s] = lB[cur][16 + 8 * h16 + s][ncol];
            }
        }

#pragma unroll
        for (int mt = 0; mt < 4; ++mt)
#pragma unroll
            for (int nt = 0; nt < 2; ++nt)
                acc[mt][nt] = __builtin_amdgcn_wmma_f32_16x16x32_f16(
                    false, a[mt].h, false, b[nt].h,
                    (short)0, acc[mt][nt], false, false);

#if USE_ASYNC_LDS
        cur ^= 1;
    }
#else
    }
#endif

    // Epilogue: lanes 0-15: N=lane, VGPR r -> M=r; lanes 16-31: N=lane-16, M=r+8
#pragma unroll
    for (int mt = 0; mt < 4; ++mt)
#pragma unroll
        for (int nt = 0; nt < 2; ++nt) {
            const int col   = n_blk + wn + nt * 16 + l15;
            const int rbase = m_blk + wm + mt * 16 + h16 * 8;
#pragma unroll
            for (int r = 0; r < 8; ++r)
                out[(size_t)(rbase + r) * N_ + col] = acc[mt][nt][r];
        }
}

// ============================================================
extern "C" void kernel_launch(void* const* d_in, const int* in_sizes, int n_in,
                              void* d_out, int out_size, void* d_ws, size_t ws_size,
                              hipStream_t stream) {
    const float* x     = (const float*)d_in[0];
    const int*   gidx  = (const int*)d_in[1];
    const float* scale = (const float*)d_in[2];
    const float* lut   = (const float*)d_in[3];
    float*       out   = (float*)d_out;

    // workspace: Wt f16 [K_][N_] (32 MB) then Xh f16 [M_][K_] (64 MB)
    _Float16* Wt = (_Float16*)d_ws;
    _Float16* Xh = (_Float16*)((char*)d_ws + (size_t)K_ * N_ * sizeof(_Float16));

    ghost_dequant_tr<<<dim3(N_ / 32, K_ / 32), 256, 0, stream>>>(gidx, scale, lut, Wt);
    ghost_cvt_f16<<<((size_t)M_ * K_) / (8 * 256), 256, 0, stream>>>(x, Xh);
    ghost_gemm_f16<<<dim3(N_ / BN, M_ / BM), 256, 0, stream>>>(Xh, Wt, out);
}
